// FullAttention_66803921322327
// MI455X (gfx1250) — matrix operand order repaired and checked by reading, not compiled, not run on hardware
//
#include <hip/hip_runtime.h>
#include <hip/hip_bf16.h>
#include <math.h>

// ---------------------------------------------------------------------------
// Full multi-head attention for MI455X (gfx1250, wave32, WMMA).
// B=2, N=4096, DIM=768, H=8, DK=DV=64.  bf16 WMMA (16x16x32) with fp32 accum,
// flash-attention (online softmax) so the N x N score matrix never hits HBM.
// GEMM k-loops are software-pipelined with an unroll-by-2 register ping-pong
// (no rotation copies): loads for chunk k+1 issue before WMMAs of chunk k.
// ---------------------------------------------------------------------------

typedef __attribute__((ext_vector_type(16))) __bf16 v16bf;
typedef __attribute__((ext_vector_type(8)))  __bf16 v8bf;
typedef __attribute__((ext_vector_type(8)))  float  v8f;

#define B_   2
#define N_   4096
#define DIM_ 768
#define H_   8
#define DK_  64
#define HD_  512   // H*DK

union BFPack { v16bf v; v8bf h[2]; };
union BF8Pack { v8bf v; __bf16 e[8]; };

__device__ __forceinline__ v16bf load16(const __bf16* p) { return *(const v16bf*)p; }
__device__ __forceinline__ v8bf  load8 (const __bf16* p) { return *(const v8bf*)p; }

__device__ __forceinline__ v8f wmma_bf16(v16bf a, v16bf b, v8f c) {
  return __builtin_amdgcn_wmma_f32_16x16x32_bf16(false, a, false, b, (short)0, c,
                                                 false, false);
}

// load one A slab (16x32, A-layout) + four B slabs (32x16 each) at k-offset k0
__device__ __forceinline__ void load_ab(const __bf16* __restrict__ arow,
                                        const __bf16* const brow[4], int k0,
                                        int half, BFPack& a, v16bf b[4]) {
  a.h[0] = load8(arow + k0 + half * 16);
  a.h[1] = load8(arow + k0 + half * 16 + 8);
#pragma unroll
  for (int t = 0; t < 4; ++t) b[t] = load16(brow[t] + k0);
}

__device__ __forceinline__ void mma4(const BFPack& a, const v16bf b[4],
                                     v8f acc[4]) {
#pragma unroll
  for (int t = 0; t < 4; ++t) acc[t] = wmma_bf16(a.v, b[t], acc[t]);
}

// ---------------------------------------------------------------------------
// fp32 -> bf16 elementwise convert
__global__ void cvt_f32_bf16(const float* __restrict__ src,
                             __bf16* __restrict__ dst, int n) {
  int i = blockIdx.x * blockDim.x + threadIdx.x;
  if (i < n) dst[i] = (__bf16)src[i];
}

// fp32 [rows x cols] -> bf16 transposed [cols x rows]
__global__ void cvt_transpose(const float* __restrict__ src,
                              __bf16* __restrict__ dst, int rows, int cols) {
  int i = blockIdx.x * blockDim.x + threadIdx.x;
  if (i < rows * cols) {
    int c = i / rows, r = i % rows;          // dst[c*rows + r] = src[r][c]
    dst[i] = (__bf16)src[r * cols + c];
  }
}

// ---------------------------------------------------------------------------
// QKV projection: [8192 x 768] x [768 x 512] per matrix; each wave does a
// 16x64 output tile.  Q is pre-scaled by 1/sqrt(DK).  V is stored transposed
// ([b,h,dv,n]) so it can be the WMMA B-operand with contiguous loads later.
__global__ __launch_bounds__(256) void qkv_proj(
    const __bf16* __restrict__ xb,
    const __bf16* __restrict__ wqT, const __bf16* __restrict__ wkT,
    const __bf16* __restrict__ wvT,
    __bf16* __restrict__ qbuf, __bf16* __restrict__ kbuf,
    __bf16* __restrict__ vtbuf) {
  const int lane = threadIdx.x & 31;
  const int wid  = blockIdx.x * 8 + (threadIdx.x >> 5);
  const int mat  = wid >> 12;             // 0:Q 1:K 2:V  (4096 waves each)
  const int rem  = wid & 4095;
  const int mt   = rem >> 3;              // 512 row tiles
  const int nt   = rem & 7;               // 8 col tiles (64 wide)
  const int mb = mt * 16, ob = nt * 64;
  const int half = lane >> 4, l = lane & 15;
  const __bf16* WT = (mat == 0) ? wqT : (mat == 1) ? wkT : wvT;

  const __bf16* arow = xb + (size_t)(mb + l) * DIM_;
  const __bf16* brow[4];
#pragma unroll
  for (int t = 0; t < 4; ++t)
    brow[t] = WT + (size_t)(ob + t * 16 + l) * DIM_ + half * 16;

  v8f acc[4] = {};

  // unroll-by-2 ping-pong software pipeline (KC = 24, even)
  BFPack a0, a1;
  v16bf b0[4], b1[4];
  load_ab(arow, brow, 0, half, a0, b0);
  for (int kc = 0; kc + 2 < DIM_ / 32; kc += 2) {
    load_ab(arow, brow, (kc + 1) * 32, half, a1, b1);
    mma4(a0, b0, acc);
    load_ab(arow, brow, (kc + 2) * 32, half, a0, b0);
    mma4(a1, b1, acc);
  }
  load_ab(arow, brow, (DIM_ / 32 - 1) * 32, half, a1, b1);
  mma4(a0, b0, acc);
  mma4(a1, b1, acc);

  // ---- store ----
  const int rm0 = mb + half * 8;                 // first of 8 rows, all same b
  const int b   = rm0 >> 12, n0 = rm0 & (N_ - 1);
  if (mat == 2) {
    // V: along the register (row) dim the transposed addresses are contiguous
    // -> pack 8 rows into one 16B store per column tile.
#pragma unroll
    for (int t = 0; t < 4; ++t) {
      const int oc = ob + t * 16 + l;
      const int h = oc >> 6, d = oc & 63;
      BF8Pack pk;
#pragma unroll
      for (int r = 0; r < 8; ++r) pk.e[r] = (__bf16)acc[t][r];
      *(v8bf*)(vtbuf + ((size_t)(b * H_ + h) * DK_ + d) * N_ + n0) = pk.v;
    }
  } else {
    __bf16* dst = (mat == 0) ? qbuf : kbuf;
    const float scale = (mat == 0) ? 0.125f : 1.0f;  // 1/sqrt(64) into Q
#pragma unroll
    for (int t = 0; t < 4; ++t) {
      const int oc = ob + t * 16 + l;
      const int h = oc >> 6, d = oc & 63;
#pragma unroll
      for (int r = 0; r < 8; ++r)
        dst[(((size_t)(b * H_ + h) * N_) + n0 + r) * DK_ + d] =
            (__bf16)(acc[t][r] * scale);
    }
  }
}

// ---------------------------------------------------------------------------
// Flash attention: one wave per 16-query tile, online softmax over key chunks
// of 64.  Per chunk: 8 WMMAs for S = Q K^T, V loads issued early (latency
// hidden under softmax VALU + LDS bounce), 8 WMMAs for P V.
__global__ __launch_bounds__(256) void flash_attn(
    const __bf16* __restrict__ qbuf, const __bf16* __restrict__ kbuf,
    const __bf16* __restrict__ vtbuf, __bf16* __restrict__ ab) {
  __shared__ __attribute__((aligned(16))) __bf16 lds[8 * 16 * 64];
  const int lane = threadIdx.x & 31;
  const int warp = threadIdx.x >> 5;
  const int wid  = blockIdx.x * 8 + warp;
  const int bh   = wid >> 8;               // b*H + h   (16 total)
  const int qt   = wid & 255;              // query tile (256 per head)
  const int half = lane >> 4, l = lane & 15;
  __bf16* plds = lds + warp * (16 * 64);

  // Q tile (16x64) in A-layout, two K=32 slabs
  const __bf16* qrow = qbuf + ((size_t)bh * N_ + qt * 16 + l) * DK_;
  BFPack qa[2];
#pragma unroll
  for (int kc = 0; kc < 2; ++kc) {
    qa[kc].h[0] = load8(qrow + kc * 32 + half * 16);
    qa[kc].h[1] = load8(qrow + kc * 32 + half * 16 + 8);
  }

  v8f acc[4] = {};
  float m[8], lsum[8];
#pragma unroll
  for (int r = 0; r < 8; ++r) { m[r] = -INFINITY; lsum[r] = 0.0f; }

  const __bf16* kbase = kbuf  + (size_t)bh * N_ * DK_;
  const __bf16* vbase = vtbuf + (size_t)bh * DK_ * N_;

  for (int j = 0; j < N_ / 64; ++j) {
    const int jb = j * 64;

    // prefetch K/V two chunks ahead (K/V are L2-resident; pull toward WGP)
    if (j + 2 < N_ / 64) {
      const int jf = jb + 128;
      __builtin_prefetch(kbase + (size_t)(jf + lane) * DK_, 0, 3);
      __builtin_prefetch(kbase + (size_t)(jf + 32 + lane) * DK_, 0, 3);
      __builtin_prefetch(vbase + (size_t)lane * N_ + jf, 0, 3);
      __builtin_prefetch(vbase + (size_t)(lane + 32) * N_ + jf, 0, 3);
    }

    // S = Q K^T for 64 keys -> four 16x16 C tiles
    v8f s[4] = {};
#pragma unroll
    for (int t = 0; t < 4; ++t) {
      const __bf16* krow = kbase + (size_t)(jb + t * 16 + l) * DK_;
      v16bf kb0 = load16(krow + half * 16);
      v16bf kb1 = load16(krow + 32 + half * 16);
      s[t] = wmma_bf16(qa[0].v, kb0, s[t]);
      s[t] = wmma_bf16(qa[1].v, kb1, s[t]);
    }

    // issue V-operand loads NOW; latency hides under softmax + LDS bounce
    v16bf vb0[4], vb1[4];
#pragma unroll
    for (int t = 0; t < 4; ++t) {
      const __bf16* vrow = vbase + (size_t)(t * 16 + l) * N_ + jb;
      vb0[t] = load16(vrow + half * 16);
      vb1[t] = load16(vrow + 32 + half * 16);
    }

    // online softmax: row max (cross-lane within 16-lane halves), rescale
    float psum[8];
#pragma unroll
    for (int r = 0; r < 8; ++r) {
      float v = fmaxf(fmaxf(s[0][r], s[1][r]), fmaxf(s[2][r], s[3][r]));
      v = fmaxf(v, __shfl_xor(v, 1, 32));
      v = fmaxf(v, __shfl_xor(v, 2, 32));
      v = fmaxf(v, __shfl_xor(v, 4, 32));
      v = fmaxf(v, __shfl_xor(v, 8, 32));
      const float mnew  = fmaxf(m[r], v);
      const float alpha = __expf(m[r] - mnew);
      m[r] = mnew;
      lsum[r] *= alpha;
#pragma unroll
      for (int t = 0; t < 4; ++t) acc[t][r] *= alpha;
      psum[r] = 0.0f;
    }

    // P = exp(S - m): accumulate row sums, stage bf16 tile row-major in LDS
#pragma unroll
    for (int t = 0; t < 4; ++t) {
#pragma unroll
      for (int r = 0; r < 8; ++r) {
        const float p = __expf(s[t][r] - m[r]);
        psum[r] += p;
        plds[(half * 8 + r) * 64 + t * 16 + l] = (__bf16)p;
      }
    }
#pragma unroll
    for (int r = 0; r < 8; ++r) {
      float v = psum[r];
      v += __shfl_xor(v, 1, 32);
      v += __shfl_xor(v, 2, 32);
      v += __shfl_xor(v, 4, 32);
      v += __shfl_xor(v, 8, 32);
      lsum[r] += v;
    }

    // cross-lane LDS store -> load hazard: LDS is in-order per wave; force
    // completion + block compiler reordering.
    asm volatile("s_wait_dscnt 0" ::: "memory");

    // P tile back in A-layout (two K=32 slabs of the 16x64 tile)
    BFPack pa[2];
#pragma unroll
    for (int kc = 0; kc < 2; ++kc) {
      pa[kc].h[0] = load8(plds + l * 64 + kc * 32 + half * 16);
      pa[kc].h[1] = load8(plds + l * 64 + kc * 32 + half * 16 + 8);
    }

    // acc += P V  (V operands already in flight / arrived)
#pragma unroll
    for (int t = 0; t < 4; ++t) {
      acc[t] = wmma_bf16(pa[0].v, vb0[t], acc[t]);
      acc[t] = wmma_bf16(pa[1].v, vb1[t], acc[t]);
    }
  }

  // normalize and emit bf16 attn output [b, n, h*64 + dv]
  const int b = bh >> 3, h = bh & 7;
#pragma unroll
  for (int r = 0; r < 8; ++r) {
    const float inv = 1.0f / lsum[r];
    const int n = qt * 16 + half * 8 + r;
#pragma unroll
    for (int t = 0; t < 4; ++t) {
      ab[((size_t)(b * N_ + n)) * HD_ + h * 64 + t * 16 + l] =
          (__bf16)(acc[t][r] * inv);
    }
  }
}

// ---------------------------------------------------------------------------
// Output projection: [8192 x 512] x [512 x 768] + bias, fp32 out.
__global__ __launch_bounds__(256) void out_proj(
    const __bf16* __restrict__ ab, const __bf16* __restrict__ woT,
    const float* __restrict__ bo, float* __restrict__ out) {
  const int lane = threadIdx.x & 31;
  const int wid  = blockIdx.x * 8 + (threadIdx.x >> 5);
  const int mt = wid / 12, nt = wid % 12;
  const int mb = mt * 16, ob = nt * 64;
  const int half = lane >> 4, l = lane & 15;

  const __bf16* arow = ab + (size_t)(mb + l) * HD_;
  const __bf16* brow[4];
#pragma unroll
  for (int t = 0; t < 4; ++t)
    brow[t] = woT + (size_t)(ob + t * 16 + l) * HD_ + half * 16;

  v8f acc[4] = {};

  // unroll-by-2 ping-pong software pipeline (KC = 16, even)
  BFPack a0, a1;
  v16bf b0[4], b1[4];
  load_ab(arow, brow, 0, half, a0, b0);
  for (int kc = 0; kc + 2 < HD_ / 32; kc += 2) {
    load_ab(arow, brow, (kc + 1) * 32, half, a1, b1);
    mma4(a0, b0, acc);
    load_ab(arow, brow, (kc + 2) * 32, half, a0, b0);
    mma4(a1, b1, acc);
  }
  load_ab(arow, brow, (HD_ / 32 - 1) * 32, half, a1, b1);
  mma4(a0, b0, acc);
  mma4(a1, b1, acc);

#pragma unroll
  for (int t = 0; t < 4; ++t) {
    const int col = ob + t * 16 + l;
    const float bias = bo[col];
#pragma unroll
    for (int r = 0; r < 8; ++r) {
      const int row = mb + half * 8 + r;
      out[(size_t)row * DIM_ + col] = acc[t][r] + bias;
    }
  }
}

// ---------------------------------------------------------------------------
extern "C" void kernel_launch(void* const* d_in, const int* in_sizes, int n_in,
                              void* d_out, int out_size, void* d_ws,
                              size_t ws_size, hipStream_t stream) {
  (void)in_sizes; (void)n_in; (void)out_size; (void)ws_size;
  const float* x  = (const float*)d_in[0];
  const float* Wq = (const float*)d_in[1];
  const float* Wk = (const float*)d_in[2];
  const float* Wv = (const float*)d_in[3];
  const float* Wo = (const float*)d_in[4];
  const float* bo = (const float*)d_in[5];
  float* out = (float*)d_out;

  // workspace layout (bf16 elements, all regions 256-elem aligned)
  __bf16* ws   = (__bf16*)d_ws;
  __bf16* xb   = ws;                      // 8192*768      = 6291456
  __bf16* wqT  = ws + 6291456;            // 512*768       = 393216
  __bf16* wkT  = ws + 6684672;
  __bf16* wvT  = ws + 7077888;
  __bf16* woT  = ws + 7471104;            // 768*512       = 393216
  __bf16* qbuf = ws + 7864320;            // 2*8*4096*64   = 4194304
  __bf16* kbuf = ws + 12058624;
  __bf16* vtbuf= ws + 16252928;           // transposed V
  __bf16* abuf = ws + 20447232;           // 8192*512 attn output
  // total = 24641536 bf16 = ~47 MB

  cvt_f32_bf16 <<<(B_ * N_ * DIM_) / 256, 256, 0, stream>>>(x, xb, B_ * N_ * DIM_);
  cvt_transpose<<<(DIM_ * HD_) / 256,     256, 0, stream>>>(Wq, wqT, DIM_, HD_);
  cvt_transpose<<<(DIM_ * HD_) / 256,     256, 0, stream>>>(Wk, wkT, DIM_, HD_);
  cvt_transpose<<<(DIM_ * HD_) / 256,     256, 0, stream>>>(Wv, wvT, DIM_, HD_);
  cvt_transpose<<<(HD_ * DIM_) / 256,     256, 0, stream>>>(Wo, woT, HD_, DIM_);

  // 3 matrices * 512 row tiles * 8 col tiles = 12288 waves = 1536 blocks
  qkv_proj  <<<1536, 256, 0, stream>>>(xb, wqT, wkT, wvT, qbuf, kbuf, vtbuf);
  // 16 (b,h) * 256 query tiles = 4096 waves = 512 blocks
  flash_attn<<<512, 256, 0, stream>>>(qbuf, kbuf, vtbuf, abuf);
  // 512 row tiles * 12 col tiles = 6144 waves = 768 blocks
  out_proj  <<<768, 256, 0, stream>>>(abuf, woT, bo, out);
}
